// Model_21543555956927
// MI455X (gfx1250) — matrix operand order repaired and checked by reading, not compiled, not run on hardware
//
#include <hip/hip_runtime.h>

// MI455X / gfx1250, wave32. 31-layer MLP (13->16, 30x 16->16, 16->1) via
// V_WMMA_F32_16X16X32_F16 (f16 inputs, f32 accumulate), H=16 == WMMA tile.
//
// Orientation: D[m][n] = sum_k W[m][k] * h[n][k]  (m=out feature, n=batch col)
//   A operand = weights  : lane -> m, elems -> k (K padded 16->32 with zeros)
//   B operand = activ.   : lane -> n, elems -> k
//   D/C       : lane -> n, VGPR r -> m = r + 8*(lane>=16)
// Layer chaining: features 8..15 for batch col n live in lane n+16 ->
// one ds_swizzle SWAPX16 per packed half2 register rebuilds the next B.
// B's K=16..31 range (elems 8..15) needs NO zeroing: A's elems 8..15 are zero
// for every lane, so those products vanish regardless of B's (finite) content.

typedef __attribute__((ext_vector_type(16))) _Float16 v16h;
typedef __attribute__((ext_vector_type(8)))  float    v8f;
typedef __attribute__((ext_vector_type(4)))  float    v4f;

#define N_HID 30
#define TILES 8   // 16-row batch tiles per wave (amortizes weight fetch 8x)

union U16 { v16h h; int i[8]; };

__device__ __forceinline__ int swap16(int v) {
    // ds_swizzle group-of-32: and=0x1f, or=0, xor=0x10 (SWAPX16)
    return __builtin_amdgcn_ds_swizzle(v, 0x401F);
}
__device__ __forceinline__ int pk2(float a, float b) {
    return __builtin_bit_cast(int, __builtin_amdgcn_cvt_pkrtz(a, b));
}

__global__ __launch_bounds__(256)
void mlp31_wmma_f16(const float* __restrict__ x,
                    const float* __restrict__ W_in,
                    const float* __restrict__ b_in,
                    const float* __restrict__ W_h,
                    const float* __restrict__ b_h,
                    const float* __restrict__ W_out,
                    const float* __restrict__ b_out,
                    float* __restrict__ y,
                    int nrows)
{
    const int lane = threadIdx.x & 31;
    const int lo   = lane & 15;   // row/col within 16x16 tile
    const int hi   = lane >> 4;   // which K/M half this lane covers
    const int wave = (int)(blockIdx.x * (blockDim.x >> 5) + (threadIdx.x >> 5));
    const int rowBase = wave * (TILES * 16);
    if (rowBase >= nrows) return;          // wave-uniform

    // ---------------- input layer: D0 = W_in(16x13) * x^T + b_in ----------------
    U16 aIn;
    {
        const float* wr = W_in + lo * 13 + hi * 8;
        float t[8];
        #pragma unroll
        for (int e = 0; e < 8; ++e) t[e] = 0.f;
        if (hi == 0) {
            #pragma unroll
            for (int e = 0; e < 8; ++e) t[e] = wr[e];      // k = 0..7
        } else {
            #pragma unroll
            for (int e = 0; e < 5; ++e) t[e] = wr[e];      // k = 8..12 (13..15 = 0)
        }
        aIn.i[0] = pk2(t[0], t[1]);
        aIn.i[1] = pk2(t[2], t[3]);
        aIn.i[2] = pk2(t[4], t[5]);
        aIn.i[3] = pk2(t[6], t[7]);
        aIn.i[4] = aIn.i[5] = aIn.i[6] = aIn.i[7] = 0;     // K 16..31 pad
    }
    v8f cIn;
    {
        const float* bp = b_in + hi * 8;
        #pragma unroll
        for (int r = 0; r < 8; ++r) cIn[r] = bp[r];
    }

    v8f d[TILES];
    #pragma unroll
    for (int j = 0; j < TILES; ++j) {
        const int row = rowBase + j * 16 + lo;
        float t[14];
        #pragma unroll
        for (int e = 0; e < 14; ++e) t[e] = 0.f;
        if (row < nrows) {                                 // low lanes k=0..7, high lanes k=8..12
            const float* xp = x + (size_t)row * 13 + hi * 8;
            #pragma unroll
            for (int e = 0; e < 8; ++e) if (hi == 0 || e < 5) t[e] = xp[e];
        }
        U16 bx;
        bx.i[0] = pk2(t[0],  t[1]);
        bx.i[1] = pk2(t[2],  t[3]);
        bx.i[2] = pk2(t[4],  t[5]);
        bx.i[3] = pk2(t[6],  t[7]);
        bx.i[4] = pk2(t[8],  t[9]);
        bx.i[5] = pk2(t[10], t[11]);
        bx.i[6] = pk2(t[12], t[13]);
        bx.i[7] = 0;
        d[j] = __builtin_amdgcn_wmma_f32_16x16x32_f16(
                   false, aIn.h, false, bx.h, (short)0, cIn, false, false);
    }

    // ---------------- 30 hidden layers: D = W_l * relu(D) + b_l ----------------
    #pragma unroll 1
    for (int l = 0; l < N_HID; ++l) {
        const float* wr = W_h + l * 256 + lo * 16 + hi * 8;   // 32B aligned
        v4f w0 = *(const v4f*)(wr);
        v4f w1 = *(const v4f*)(wr + 4);
        U16 a;
        a.i[0] = pk2(w0[0], w0[1]);
        a.i[1] = pk2(w0[2], w0[3]);
        a.i[2] = pk2(w1[0], w1[1]);
        a.i[3] = pk2(w1[2], w1[3]);
        a.i[4] = a.i[5] = a.i[6] = a.i[7] = 0;                // K 16..31 pad

        const float* bp = b_h + l * 16 + hi * 8;              // 32B aligned
        v4f bb0 = *(const v4f*)(bp);
        v4f bb1 = *(const v4f*)(bp + 4);
        v8f c;
        c[0] = bb0[0]; c[1] = bb0[1]; c[2] = bb0[2]; c[3] = bb0[3];
        c[4] = bb1[0]; c[5] = bb1[1]; c[6] = bb1[2]; c[7] = bb1[3];

        #pragma unroll
        for (int j = 0; j < TILES; ++j) {
            // relu + pack own features (m = 0..7 low lanes / 8..15 high lanes)
            int p0 = pk2(fmaxf(d[j][0], 0.f), fmaxf(d[j][1], 0.f));
            int p1 = pk2(fmaxf(d[j][2], 0.f), fmaxf(d[j][3], 0.f));
            int p2 = pk2(fmaxf(d[j][4], 0.f), fmaxf(d[j][5], 0.f));
            int p3 = pk2(fmaxf(d[j][6], 0.f), fmaxf(d[j][7], 0.f));
            // pull features 8..15 for this batch col from partner lane (+16)
            int q0 = swap16(p0);
            int q1 = swap16(p1);
            int q2 = swap16(p2);
            int q3 = swap16(p3);
            U16 b;
            b.i[0] = p0;   // k = 0..7   (low lanes)  / k = 8..15 (high lanes)
            b.i[1] = p1;
            b.i[2] = p2;
            b.i[3] = p3;
            b.i[4] = q0;   // K 16..31 pad: don't-care in all lanes (A's zeros)
            b.i[5] = q1;
            b.i[6] = q2;
            b.i[7] = q3;   // high-lane garbage is finite and killed by A's zeros
            d[j] = __builtin_amdgcn_wmma_f32_16x16x32_f16(
                       false, a.h, false, b.h, (short)0, c, false, false);
        }
    }

    // ---------------- output layer: y[n] = relu(D)[.,n] . W_out + b_out ----------------
    v4f wo0 = *(const v4f*)(W_out + hi * 8);
    v4f wo1 = *(const v4f*)(W_out + hi * 8 + 4);
    const float bo = b_out[0];
    #pragma unroll
    for (int j = 0; j < TILES; ++j) {
        float acc = fmaxf(d[j][0], 0.f) * wo0[0];
        acc += fmaxf(d[j][1], 0.f) * wo0[1];
        acc += fmaxf(d[j][2], 0.f) * wo0[2];
        acc += fmaxf(d[j][3], 0.f) * wo0[3];
        acc += fmaxf(d[j][4], 0.f) * wo1[0];
        acc += fmaxf(d[j][5], 0.f) * wo1[1];
        acc += fmaxf(d[j][6], 0.f) * wo1[2];
        acc += fmaxf(d[j][7], 0.f) * wo1[3];
        // partner lane holds the other 8 features for the same batch col
        float other = __builtin_bit_cast(float, swap16(__builtin_bit_cast(int, acc)));
        float tot = acc + other + bo;
        const int row = rowBase + j * 16 + lo;
        if (hi == 0 && row < nrows) y[row] = tot;
    }
}

extern "C" void kernel_launch(void* const* d_in, const int* in_sizes, int n_in,
                              void* d_out, int out_size, void* d_ws, size_t ws_size,
                              hipStream_t stream) {
    const float* x     = (const float*)d_in[0];
    const float* W_in  = (const float*)d_in[1];
    const float* b_in  = (const float*)d_in[2];
    const float* W_h   = (const float*)d_in[3];
    const float* b_h   = (const float*)d_in[4];
    const float* W_out = (const float*)d_in[5];
    const float* b_out = (const float*)d_in[6];
    float* y = (float*)d_out;

    const int nrows = in_sizes[0] / 13;                  // 2,097,152
    const int rowsPerBlock = (256 / 32) * TILES * 16;    // 8 waves * 128 rows = 1024
    dim3 block(256);
    dim3 grid((nrows + rowsPerBlock - 1) / rowsPerBlock);
    hipLaunchKernelGGL(mlp31_wmma_f16, grid, block, 0, stream,
                       x, W_in, b_in, W_h, b_h, W_out, b_out, y, nrows);
}
